// BulkFormer_63324997812499
// MI455X (gfx1250) — matrix operand order repaired and checked by reading, not compiled, not run on hardware
//
#include <hip/hip_runtime.h>
#include <hip/hip_bf16.h>
#include <stdint.h>

#define DIMF   256
#define G_GENES 20000
#define BATCH  2
#define NEDGE  200000
#define BINSN  10
#define HEADSN 4
#define TOKS   (BATCH * G_GENES)   // 40000
#define SORTP  32768               // padded pow2 for bitonic sort of G
#define LDA_P  40                  // padded LDS row stride (halves) to avoid bank conflicts

typedef __attribute__((ext_vector_type(8)))  __bf16 bf8;
typedef __attribute__((ext_vector_type(16))) __bf16 bf16x16;
typedef __attribute__((ext_vector_type(8)))  float  f32x8;

// ---------------------------------------------------------------- helpers
__device__ __forceinline__ unsigned short f2bf(float f) {
  union { float f; unsigned int u; } v; v.f = f;
  unsigned int u = v.u;
  unsigned int r = u + 0x7FFFu + ((u >> 16) & 1u);   // RNE
  return (unsigned short)(r >> 16);
}

__device__ __forceinline__ unsigned int pack2(float a, float b) {
  return (unsigned int)f2bf(a) | ((unsigned int)f2bf(b) << 16);
}

// Build a v16bf fragment from two 16B LDS chunks (chunk1 at p, chunk2 at p+gap halves)
__device__ __forceinline__ bf16x16 load_frag(const unsigned short* p, int gap) {
  bf8 lo = *(const bf8*)p;
  bf8 hi = *(const bf8*)(p + gap);
  bf16x16 r;
#pragma unroll
  for (int i = 0; i < 8; ++i) { r[i] = lo[i]; r[i + 8] = hi[i]; }
  return r;
}

// CDNA5 async global->LDS copy (ASYNCcnt) -- bypasses VGPRs for the weight tiles.
// Low 32 bits of a generic LDS pointer are the LDS byte offset (ISA aperture rule).
__device__ __forceinline__ void async_b128(void* ldsPtr, const void* g) {
  unsigned off = (unsigned)(size_t)ldsPtr;
  asm volatile("global_load_async_to_lds_b128 %0, %1, off"
               :: "v"(off), "v"(g) : "memory");
}
__device__ __forceinline__ void wait_async0() {
  asm volatile("s_wait_asynccnt 0x0" ::: "memory");
}

// ---------------------------------------------------------------- weight prep: f32 [K][N] -> bf16 [N][K]
__global__ void prep_wt(const float* __restrict__ W, unsigned short* __restrict__ Wt,
                        int K, int N) {
  size_t i = (size_t)blockIdx.x * 256 + threadIdx.x;
  if (i >= (size_t)K * N) return;
  int k = (int)(i / N), n = (int)(i % N);
  Wt[(size_t)n * K + k] = f2bf(W[i]);
}

// ---------------------------------------------------------------- main WMMA GEMM
// C[M,N] = act(A[M,K] * Wt^T + bias) (+ R).  Wt is bf16 [N][K].  N%128==0, K%32==0.
// 128x128 block tile, BK=32, double-buffered LDS, async-to-LDS weight streaming.
template<int ACT, bool RESID>
__global__ __launch_bounds__(256)
void gemm_bf16(const float* __restrict__ A, const unsigned short* __restrict__ Wt,
               const float* __restrict__ bias, const float* __restrict__ R,
               float* __restrict__ C, int M, int N, int K) {
  __shared__ alignas(32) unsigned short lA[2][128 * LDA_P];
  __shared__ alignas(32) unsigned short lB[2][128 * LDA_P];
  const int m0 = blockIdx.x * 128;
  const int n0 = blockIdx.y * 128;
  const int t = threadIdx.x;
  const int w = t >> 5, l = t & 31;
  const int wm = (w >> 1) * 32;      // 0..96 (4 M-waves)
  const int wn = (w & 1) * 64;       // 0 | 64 (2 N-waves)
  const int row = t >> 1;            // 0..127 (coop load row)
  const int c0 = (t & 1) * 16;       // element offset within 32-wide K slice
  f32x8 acc[2][4] = {};

  auto loadA = [&](int kb, int buf) {
    int gr = m0 + row;
    float4 a0, a1, a2, a3;
    if (gr < M) {
      const float4* p = (const float4*)(A + (size_t)gr * K + kb + c0);
      a0 = p[0]; a1 = p[1]; a2 = p[2]; a3 = p[3];
    } else {
      a0 = make_float4(0.f, 0.f, 0.f, 0.f); a1 = a0; a2 = a0; a3 = a0;
    }
    uint4 p0 = make_uint4(pack2(a0.x, a0.y), pack2(a0.z, a0.w),
                          pack2(a1.x, a1.y), pack2(a1.z, a1.w));
    uint4 p1 = make_uint4(pack2(a2.x, a2.y), pack2(a2.z, a2.w),
                          pack2(a3.x, a3.y), pack2(a3.z, a3.w));
    *(uint4*)&lA[buf][row * LDA_P + c0]     = p0;
    *(uint4*)&lA[buf][row * LDA_P + c0 + 8] = p1;
  };
  auto loadB = [&](int kb, int buf) {     // bf16 weights stream straight to LDS
    const unsigned short* g = Wt + (size_t)(n0 + row) * K + kb + c0;
    async_b128(&lB[buf][row * LDA_P + c0],     g);
    async_b128(&lB[buf][row * LDA_P + c0 + 8], g + 8);
  };

  loadA(0, 0);
  loadB(0, 0);
  int cur = 0;
  for (int kb = 0; kb < K; kb += 32) {
    wait_async0();
    __syncthreads();                      // buf[cur] ready; buf[cur^1] free
    if (kb + 32 < K) { loadA(kb + 32, cur ^ 1); loadB(kb + 32, cur ^ 1); }

    const int aK = (l >> 4) * 8;
    const int bK = (l >> 4) * 8;
    bf16x16 af[2], bfr[4];
#pragma unroll
    for (int mt = 0; mt < 2; ++mt)
      af[mt] = load_frag(&lA[cur][(wm + mt * 16 + (l & 15)) * LDA_P + aK], 16);
#pragma unroll
    for (int nt = 0; nt < 4; ++nt)
      bfr[nt] = load_frag(&lB[cur][(wn + nt * 16 + (l & 15)) * LDA_P + bK], 16);
#pragma unroll
    for (int mt = 0; mt < 2; ++mt)
#pragma unroll
      for (int nt = 0; nt < 4; ++nt)
        acc[mt][nt] = __builtin_amdgcn_wmma_f32_16x16x32_bf16(
            false, af[mt], false, bfr[nt], (short)0, acc[mt][nt], false, false);
    cur ^= 1;
  }

  // --- epilogue: bias + act (+ residual)
#pragma unroll
  for (int mt = 0; mt < 2; ++mt) {
    int rbase = m0 + wm + mt * 16 + ((l >> 4) * 8);
#pragma unroll
    for (int nt = 0; nt < 4; ++nt) {
      int col = n0 + wn + nt * 16 + (l & 15);
      float bval = bias ? bias[col] : 0.f;
#pragma unroll
      for (int j = 0; j < 8; ++j) {
        int r = rbase + j;
        if (r < M) {
          float vv = acc[mt][nt][j] + bval;
          if (ACT == 1) vv = fmaxf(vv, 0.f);
          else if (ACT == 2) {
            float x3 = vv * vv * vv;
            vv = 0.5f * vv * (1.f + tanhf(0.7978845608f * (vv + 0.044715f * x3)));
          } else if (ACT == 3) {
            vv = vv > 0.f ? vv + 1.f : __expf(vv);   // elu(x)+1
          }
          if (RESID) vv += R[(size_t)r * N + col];
          C[(size_t)r * N + col] = vv;
        }
      }
    }
  }
}

// ---------------------------------------------------------------- Performer kv = K^T V (64x64 per seq*head), split-K atomics
__global__ __launch_bounds__(256)
void kv_wmma(const float* __restrict__ Kmat, const float* __restrict__ Vmat,
             float* __restrict__ kvOut, int n, int chunk) {
  __shared__ alignas(32) unsigned short lKT[64 * LDA_P];   // [d][tok]
  __shared__ alignas(32) unsigned short lVT[64 * LDA_P];   // [e][tok]
  const int sh = blockIdx.x;
  const int s = sh / HEADSN, h = sh % HEADSN;
  const int t0 = blockIdx.y * chunk;
  int t1 = t0 + chunk; if (t1 > n) t1 = n;
  const int t = threadIdx.x, w = t >> 5, l = t & 31;
  const int wm = (w >> 1) * 16;   // d tile
  const int wn = (w & 1) * 32;    // e tile (x2)
  const size_t rowBase = (size_t)s * n;
  f32x8 acc[2] = {};

  for (int kb0 = t0; kb0 < t1; kb0 += 32) {
    int tok = t >> 3;
    int d0  = (t & 7) * 8;
    int gt  = kb0 + tok;
    float4 a0, a1, b0, b1;
    if (gt < t1) {
      const float* kp = Kmat + (rowBase + gt) * DIMF + h * 64 + d0;
      const float* vp = Vmat + (rowBase + gt) * DIMF + h * 64 + d0;
      a0 = *(const float4*)kp; a1 = *(const float4*)(kp + 4);
      b0 = *(const float4*)vp; b1 = *(const float4*)(vp + 4);
    } else { a0 = make_float4(0.f,0.f,0.f,0.f); a1 = a0; b0 = a0; b1 = a0; }
    unsigned short ks[8] = { f2bf(a0.x),f2bf(a0.y),f2bf(a0.z),f2bf(a0.w),
                             f2bf(a1.x),f2bf(a1.y),f2bf(a1.z),f2bf(a1.w) };
    unsigned short vs[8] = { f2bf(b0.x),f2bf(b0.y),f2bf(b0.z),f2bf(b0.w),
                             f2bf(b1.x),f2bf(b1.y),f2bf(b1.z),f2bf(b1.w) };
#pragma unroll
    for (int i = 0; i < 8; ++i) {
      lKT[(d0 + i) * LDA_P + tok] = ks[i];
      lVT[(d0 + i) * LDA_P + tok] = vs[i];
    }
    __syncthreads();

    bf16x16 af = load_frag(&lKT[(wm + (l & 15)) * LDA_P + ((l >> 4) * 8)], 16);
    int bK = (l >> 4) * 8;
#pragma unroll
    for (int nt = 0; nt < 2; ++nt) {
      bf16x16 bv = load_frag(&lVT[(wn + nt * 16 + (l & 15)) * LDA_P + bK], 16);
      acc[nt] = __builtin_amdgcn_wmma_f32_16x16x32_bf16(
          false, af, false, bv, (short)0, acc[nt], false, false);
    }
    __syncthreads();
  }

  float* out = kvOut + (size_t)sh * 64 * 64;
#pragma unroll
  for (int nt = 0; nt < 2; ++nt)
#pragma unroll
    for (int j = 0; j < 8; ++j) {
      int d = wm + j + ((l >> 4) * 8);
      int e = wn + nt * 16 + (l & 15);
      atomicAdd(&out[d * 64 + e], acc[nt][j]);
    }
}

// ---------------------------------------------------------------- k column-sum (split with atomics)
__global__ void colsum_kernel(const float* __restrict__ Kmat, float* __restrict__ KS,
                              int n, int chunk) {
  __shared__ float red[256];
  int sh = blockIdx.x;
  int s = sh / HEADSN, h = sh % HEADSN;
  int c0 = blockIdx.y * chunk;
  int c1 = c0 + chunk; if (c1 > n) c1 = n;
  int d = threadIdx.x & 63, part = threadIdx.x >> 6;
  float sum = 0.f;
  for (int i = c0 + part; i < c1; i += 4)
    sum += Kmat[((size_t)s * n + i) * DIMF + h * 64 + d];
  red[threadIdx.x] = sum;
  __syncthreads();
  if (part == 0)
    atomicAdd(&KS[sh * 64 + d], red[d] + red[64 + d] + red[128 + d] + red[192 + d]);
}

// ---------------------------------------------------------------- att = (q . kv) / (q . ksum + eps)
__global__ __launch_bounds__(256)
void att_kernel(const float* __restrict__ Q, const float* __restrict__ KV,
                const float* __restrict__ KS, float* __restrict__ ATT, int n) {
  __shared__ float lkv[64 * 64];
  __shared__ float lks[64];
  int tok0 = blockIdx.x * 4;
  int s = tok0 / n;
  int tok = tok0 + (threadIdx.x >> 6);
  int e = threadIdx.x & 63;
  for (int h = 0; h < HEADSN; ++h) {
    const float* kv = KV + ((size_t)(s * HEADSN + h)) * 4096;
    for (int i = threadIdx.x; i < 4096; i += 256) lkv[i] = kv[i];
    if (threadIdx.x < 64) lks[threadIdx.x] = KS[(s * HEADSN + h) * 64 + threadIdx.x];
    __syncthreads();
    const float* q = Q + (size_t)tok * DIMF + h * 64;
    float num = 0.f, den = 0.f;
#pragma unroll 8
    for (int d = 0; d < 64; ++d) {
      float qd = q[d];
      num += qd * lkv[d * 64 + e];
      den += qd * lks[d];
    }
    ATT[(size_t)tok * DIMF + h * 64 + e] = num / (den + 1e-6f);
    __syncthreads();
  }
}

// ---------------------------------------------------------------- LayerNorm (wave per 256-row)
__global__ void ln_kernel(const float* __restrict__ X, const float* __restrict__ g,
                          const float* __restrict__ b, float* __restrict__ Y, int T) {
  int row = blockIdx.x * 8 + (threadIdx.x >> 5);
  if (row >= T) return;
  int l = threadIdx.x & 31;
  const float* x = X + (size_t)row * DIMF;
  float vals[8], s = 0.f, s2 = 0.f;
#pragma unroll
  for (int i = 0; i < 8; ++i) { float v = x[l + i * 32]; vals[i] = v; s += v; s2 += v * v; }
#pragma unroll
  for (int off = 16; off; off >>= 1) { s += __shfl_xor(s, off, 32); s2 += __shfl_xor(s2, off, 32); }
  float mean = s * (1.f / DIMF);
  float var = s2 * (1.f / DIMF) - mean * mean;
  float inv = rsqrtf(var + 1e-5f);
  float* y = Y + (size_t)row * DIMF;
#pragma unroll
  for (int i = 0; i < 8; ++i) { int c = l + i * 32; y[c] = (vals[i] - mean) * inv * g[c] + b[c]; }
}

// ---------------------------------------------------------------- N=1 projection (wave per row)
template<int ACT>
__global__ void gemv1(const float* __restrict__ X, const float* __restrict__ w,
                      const float* __restrict__ bias, float* __restrict__ out,
                      int T, int K) {
  int row = blockIdx.x * 8 + (threadIdx.x >> 5);
  if (row >= T) return;
  int l = threadIdx.x & 31;
  const float* x = X + (size_t)row * K;
  float s = 0.f;
  for (int i = l; i < K; i += 32) s += x[i] * w[i];
#pragma unroll
  for (int off = 16; off; off >>= 1) s += __shfl_xor(s, off, 32);
  if (l == 0) {
    s += bias[0];
    if (ACT == 1) s = fmaxf(s, 0.f);
    out[row] = s;
  }
}

// ---------------------------------------------------------------- misc elementwise
__global__ void zero_kernel(float* p, size_t n) {
  size_t i = (size_t)blockIdx.x * 256 + threadIdx.x;
  if (i < n) p[i] = 0.f;
}

__global__ void embed_kernel(const float* __restrict__ x, const float* __restrict__ ge,
                             const float* __restrict__ ae, float* __restrict__ H) {
  size_t i = (size_t)blockIdx.x * 256 + threadIdx.x;
  if (i >= (size_t)TOKS * 128) return;
  int row = (int)(i >> 7);
  int c = (int)(i & 127);
  int b = row / G_GENES, g = row % G_GENES;
  float xv = x[row];
  bool mask = (xv == -10.0f);
  float invf = __expf(-((float)(2 * c) / 256.f) * 4.60517018598809f); // 1/100^(2c/256)
  float ang = xv * invf;
  float sv = mask ? 0.f : __sinf(ang);
  float cv = mask ? 0.f : __cosf(ang);
  float* hrow = H + (size_t)row * DIMF;
  hrow[c]       = sv + ge[g * DIMF + c]       + ae[b * DIMF + c];
  hrow[c + 128] = cv + ge[g * DIMF + c + 128] + ae[b * DIMF + c + 128];
}

// ae path: x[B,G] @ ae1[G,1024] split-K with atomics
__global__ void ae1_kernel(const float* __restrict__ x, const float* __restrict__ w,
                           float* __restrict__ out) {
  int b = blockIdx.x;
  int k0 = blockIdx.y * 400, k1 = k0 + 400;
  for (int cc = 0; cc < 4; ++cc) {
    int col = threadIdx.x + cc * 256;
    float s = 0.f;
    for (int k = k0; k < k1; ++k) s += x[b * G_GENES + k] * w[(size_t)k * 1024 + col];
    atomicAdd(&out[b * 1024 + col], s);
  }
}

__global__ void bias_relu_kernel(float* v, const float* b, int n, int width) {
  int i = blockIdx.x * 256 + threadIdx.x;
  if (i < n) v[i] = fmaxf(v[i] + b[i % width], 0.f);
}

__global__ void ae2_kernel(const float* __restrict__ a1, const float* __restrict__ w,
                           const float* __restrict__ b, float* __restrict__ out) {
  int bb = blockIdx.x, c = threadIdx.x;
  float s = 0.f;
  for (int k = 0; k < 1024; ++k) s += a1[bb * 1024 + k] * w[(size_t)k * DIMF + c];
  out[bb * DIMF + c] = fmaxf(s + b[c], 0.f);
}

// ---------------------------------------------------------------- GCN
__global__ void deg_kernel(const int* __restrict__ dst, float* __restrict__ deg) {
  int e = blockIdx.x * 256 + threadIdx.x;
  if (e < NEDGE) atomicAdd(&deg[dst[e]], 1.f);
}
__global__ void dinv_kernel(float* deg) {
  int g = blockIdx.x * 256 + threadIdx.x;
  if (g < G_GENES) deg[g] = rsqrtf(fmaxf(deg[g], 1.f));
}
__global__ void gcn_scatter(const int* __restrict__ src, const int* __restrict__ dst,
                            const float* __restrict__ dinv, const float* __restrict__ Hlin,
                            float* __restrict__ agg) {
  size_t i = (size_t)blockIdx.x * 256 + threadIdx.x;
  if (i >= (size_t)NEDGE * 64) return;
  int e = (int)(i >> 6);
  int c0 = ((int)i & 63) * 4;
  int s = src[e], d = dst[e];
  float norm = dinv[s] * dinv[d];
  for (int b = 0; b < BATCH; ++b) {
    const float* hp = Hlin + ((size_t)b * G_GENES + s) * DIMF + c0;
    float* ap = agg + ((size_t)b * G_GENES + d) * DIMF + c0;
    atomicAdd(&ap[0], hp[0] * norm);
    atomicAdd(&ap[1], hp[1] * norm);
    atomicAdd(&ap[2], hp[2] * norm);
    atomicAdd(&ap[3], hp[3] * norm);
  }
}
__global__ void add_agg_kernel(float* __restrict__ X, const float* __restrict__ agg,
                               const float* __restrict__ bias) {
  size_t i = (size_t)blockIdx.x * 256 + threadIdx.x;
  if (i < (size_t)TOKS * DIMF) X[i] += agg[i] + bias[i & (DIMF - 1)];
}

// ---------------------------------------------------------------- descending bitonic argsort
__global__ void sort_init(const float* __restrict__ score, float* __restrict__ keys,
                          int* __restrict__ idx) {
  int i = blockIdx.x * 256 + threadIdx.x;
  if (i >= BATCH * SORTP) return;
  int b = i / SORTP, j = i % SORTP;
  keys[i] = (j < G_GENES) ? score[b * G_GENES + j] : -3.0e38f;
  idx[i] = j;
}
__global__ void bitonic_step(float* __restrict__ keys, int* __restrict__ idx, int j, int k) {
  int i = blockIdx.x * 256 + threadIdx.x;
  int b = i / SORTP, t = i % SORTP;
  int l = t ^ j;
  if (l <= t) return;
  size_t bi = (size_t)b * SORTP;
  float kt = keys[bi + t], kl = keys[bi + l];
  bool up = (t & k) == 0;            // descending regions
  bool sw = up ? (kt < kl) : (kt > kl);
  if (sw) {
    keys[bi + t] = kl; keys[bi + l] = kt;
    int it = idx[bi + t]; idx[bi + t] = idx[bi + l]; idx[bi + l] = it;
  }
}

__global__ void gather_kernel(const float* __restrict__ X, const int* __restrict__ ord,
                              float* __restrict__ Y) {
  size_t i = (size_t)blockIdx.x * 256 + threadIdx.x;
  if (i >= (size_t)TOKS * 64) return;
  int row = (int)(i >> 6), c = ((int)i & 63) * 4;
  int b = row / G_GENES, r = row % G_GENES;
  int sg = ord[b * SORTP + r];
  *(float4*)(Y + (size_t)row * DIMF + c) =
      *(const float4*)(X + ((size_t)b * G_GENES + sg) * DIMF + c);
}
__global__ void scatter_kernel(const float* __restrict__ Y, const int* __restrict__ ord,
                               float* __restrict__ X) {
  size_t i = (size_t)blockIdx.x * 256 + threadIdx.x;
  if (i >= (size_t)TOKS * 64) return;
  int row = (int)(i >> 6), c = ((int)i & 63) * 4;
  int b = row / G_GENES, r = row % G_GENES;
  int sg = ord[b * SORTP + r];
  *(float4*)(X + ((size_t)b * G_GENES + sg) * DIMF + c) =
      *(const float4*)(Y + (size_t)row * DIMF + c);
}

// ================================================================ host side
struct PerfW {
  const float *ln1g, *ln1b, *qbias, *kbias, *vbias, *obias, *ln2g, *ln2b, *f1b, *f2b;
  const unsigned short *qw, *kw, *vw, *ow, *f1w, *f2w;
};

static void gemm_launch(const float* A, const unsigned short* Wt, const float* bias,
                        const float* R, float* C, int M, int N, int K, int act,
                        hipStream_t stream) {
  dim3 g((M + 127) / 128, N / 128);
  dim3 b(256);
  if (R) {
    gemm_bf16<0, true><<<g, b, 0, stream>>>(A, Wt, bias, R, C, M, N, K);
  } else {
    switch (act) {
      case 1: gemm_bf16<1, false><<<g, b, 0, stream>>>(A, Wt, bias, nullptr, C, M, N, K); break;
      case 2: gemm_bf16<2, false><<<g, b, 0, stream>>>(A, Wt, bias, nullptr, C, M, N, K); break;
      case 3: gemm_bf16<3, false><<<g, b, 0, stream>>>(A, Wt, bias, nullptr, C, M, N, K); break;
      default: gemm_bf16<0, false><<<g, b, 0, stream>>>(A, Wt, bias, nullptr, C, M, N, K); break;
    }
  }
}

static void run_performer(float* X, int S, int n, const PerfW& w, float* HN,
                          float* BQ, float* BK, float* BV, float* BATT, float* TMP1,
                          float* KVb, float* KSb, hipStream_t stream) {
  int T = S * n;
  ln_kernel<<<(T + 7) / 8, 256, 0, stream>>>(X, w.ln1g, w.ln1b, HN, T);
  gemm_launch(HN, w.qw, w.qbias, nullptr, BQ, T, DIMF, DIMF, 3, stream);   // elu+1
  gemm_launch(HN, w.kw, w.kbias, nullptr, BK, T, DIMF, DIMF, 3, stream);   // elu+1
  gemm_launch(HN, w.vw, w.vbias, nullptr, BV, T, DIMF, DIMF, 0, stream);
  size_t kvN = (size_t)S * HEADSN * 64 * 64, ksN = (size_t)S * HEADSN * 64;
  zero_kernel<<<(unsigned)((kvN + 255) / 256), 256, 0, stream>>>(KVb, kvN);
  zero_kernel<<<(unsigned)((ksN + 255) / 256), 256, 0, stream>>>(KSb, ksN);
  kv_wmma<<<dim3(S * HEADSN, (n + 511) / 512), 256, 0, stream>>>(BK, BV, KVb, n, 512);
  colsum_kernel<<<dim3(S * HEADSN, (n + 2047) / 2048), 256, 0, stream>>>(BK, KSb, n, 2048);
  att_kernel<<<T / 4, 256, 0, stream>>>(BQ, KVb, KSb, BATT, n);
  gemm_launch(BATT, w.ow, w.obias, X, X, T, DIMF, DIMF, 0, stream);        // residual
  ln_kernel<<<(T + 7) / 8, 256, 0, stream>>>(X, w.ln2g, w.ln2b, HN, T);
  gemm_launch(HN, w.f1w, w.f1b, nullptr, TMP1, T, 4 * DIMF, DIMF, 2, stream); // gelu
  gemm_launch(TMP1, w.f2w, w.f2b, X, X, T, DIMF, 4 * DIMF, 0, stream);     // residual
}

extern "C" void kernel_launch(void* const* d_in, const int* in_sizes, int n_in,
                              void* d_out, int out_size, void* d_ws, size_t ws_size,
                              hipStream_t stream) {
  (void)in_sizes; (void)n_in; (void)out_size; (void)ws_size;
  auto F = [&](int i) { return (const float*)d_in[i]; };
  const float* x_in = F(0);
  const int* e_src = (const int*)d_in[1];
  const int* e_dst = e_src + NEDGE;

  // ---- workspace bump allocator
  char* ws = (char*)d_ws;
  size_t cur = 0;
  auto alloc = [&](size_t bytes) -> char* {
    char* p = ws + cur;
    cur = (cur + bytes + 255) & ~(size_t)255;
    return p;
  };
  float* H    = (float*)alloc((size_t)TOKS * DIMF * 4);
  float* HN   = (float*)alloc((size_t)TOKS * DIMF * 4);
  float* TMP1 = (float*)alloc((size_t)TOKS * 4 * DIMF * 4);
  float* BQ   = (float*)alloc((size_t)TOKS * DIMF * 4);
  float* BK   = (float*)alloc((size_t)TOKS * DIMF * 4);
  float* BV   = (float*)alloc((size_t)TOKS * DIMF * 4);
  float* BATT = (float*)alloc((size_t)TOKS * DIMF * 4);
  float* XS   = (float*)alloc((size_t)TOKS * DIMF * 4);
  float* AGG  = (float*)alloc((size_t)TOKS * DIMF * 4);
  float* GE   = (float*)alloc((size_t)G_GENES * DIMF * 4);
  float* AE1  = (float*)alloc((size_t)BATCH * 1024 * 4);
  float* AE   = (float*)alloc((size_t)BATCH * DIMF * 4);
  float* DEG  = (float*)alloc((size_t)G_GENES * 4);
  float* SCORE= (float*)alloc((size_t)TOKS * 4);
  float* KEYS = (float*)alloc((size_t)BATCH * SORTP * 4);
  int*   IDX  = (int*)  alloc((size_t)BATCH * SORTP * 4);
  float* KVb  = (float*)alloc((size_t)BATCH * BINSN * HEADSN * 4096 * 4);
  float* KSb  = (float*)alloc((size_t)BATCH * BINSN * HEADSN * 64 * 4);

  auto prep = [&](int idx, int K, int N) -> unsigned short* {
    unsigned short* dst = (unsigned short*)alloc((size_t)K * N * 2);
    size_t n = (size_t)K * N;
    prep_wt<<<(unsigned)((n + 255) / 256), 256, 0, stream>>>(F(idx), dst, K, N);
    return dst;
  };

  // ---- param leaf indices (dict insertion order)
  unsigned short* gep1t = prep(3, 256, 1024);
  unsigned short* gep2t = prep(5, 1024, 256);
  unsigned short* xp1t  = prep(7, 256, 1024);
  unsigned short* xp2t  = prep(9, 1024, 256);
  unsigned short* h1t   = prep(17, 256, 1024);

  unsigned short* gcnW[3];
  PerfW bperf[3], fperf[3];
  const float *gbLng[3], *gbLnb[3], *gcnB[3], *scW[3], *scB[3];
  for (int i = 0; i < 3; ++i) {
    int base = 21 + 38 * i;
    gbLng[i] = F(base + 0); gbLnb[i] = F(base + 1);
    gcnW[i] = prep(base + 2, 256, 256);
    gcnB[i] = F(base + 3);
    scW[i] = F(base + 4); scB[i] = F(base + 5);
    for (int pp = 0; pp < 2; ++pp) {
      int pb = base + 6 + 16 * pp;
      PerfW& w = pp == 0 ? bperf[i] : fperf[i];
      w.ln1g = F(pb + 0);  w.ln1b = F(pb + 1);
      w.qw = prep(pb + 2, 256, 256);   w.qbias = F(pb + 3);
      w.kw = prep(pb + 4, 256, 256);   w.kbias = F(pb + 5);
      w.vw = prep(pb + 6, 256, 256);   w.vbias = F(pb + 7);
      w.ow = prep(pb + 8, 256, 256);   w.obias = F(pb + 9);
      w.ln2g = F(pb + 10); w.ln2b = F(pb + 11);
      w.f1w = prep(pb + 12, 256, 1024); w.f1b = F(pb + 13);
      w.f2w = prep(pb + 14, 1024, 256); w.f2b = F(pb + 15);
    }
  }

  // ---- ae path: ae = relu(relu(x@ae1+b)@ae2+b)
  zero_kernel<<<(BATCH * 1024 + 255) / 256, 256, 0, stream>>>(AE1, (size_t)BATCH * 1024);
  ae1_kernel<<<dim3(BATCH, 50), 256, 0, stream>>>(x_in, F(11), AE1);
  bias_relu_kernel<<<(BATCH * 1024 + 255) / 256, 256, 0, stream>>>(AE1, F(12), BATCH * 1024, 1024);
  ae2_kernel<<<BATCH, 256, 0, stream>>>(AE1, F(13), F(14), AE);

  // ---- gene path: ge = gep2(relu(gep1(gene_emb)))
  gemm_launch(F(2), gep1t, F(4), nullptr, TMP1, G_GENES, 1024, 256, 1, stream);
  gemm_launch(TMP1, gep2t, F(6), nullptr, GE, G_GENES, 256, 1024, 0, stream);

  // ---- sinusoidal embedding + ge + ae
  embed_kernel<<<(unsigned)(((size_t)TOKS * 128 + 255) / 256), 256, 0, stream>>>(x_in, GE, AE, H);

  // ---- h = xp2(relu(xp1(h)))
  gemm_launch(H, xp1t, F(8), nullptr, TMP1, TOKS, 1024, 256, 1, stream);
  gemm_launch(TMP1, xp2t, F(10), nullptr, H, TOKS, 256, 1024, 0, stream);

  // ---- GCN degree normalization (edges fixed across blocks)
  zero_kernel<<<(G_GENES + 255) / 256, 256, 0, stream>>>(DEG, (size_t)G_GENES);
  deg_kernel<<<(NEDGE + 255) / 256, 256, 0, stream>>>(e_dst, DEG);
  dinv_kernel<<<(G_GENES + 255) / 256, 256, 0, stream>>>(DEG);

  // ---- 3x gbformer blocks
  for (int i = 0; i < 3; ++i) {
    ln_kernel<<<(TOKS + 7) / 8, 256, 0, stream>>>(H, gbLng[i], gbLnb[i], H, TOKS);
    // GCN: Hlin = H @ Wgcn ; agg = scatter(norm * Hlin) ; H += agg + b
    gemm_launch(H, gcnW[i], nullptr, nullptr, HN, TOKS, 256, 256, 0, stream);
    zero_kernel<<<(unsigned)(((size_t)TOKS * DIMF + 255) / 256), 256, 0, stream>>>(AGG, (size_t)TOKS * DIMF);
    gcn_scatter<<<(unsigned)(((size_t)NEDGE * 64 + 255) / 256), 256, 0, stream>>>(e_src, e_dst, DEG, HN, AGG);
    add_agg_kernel<<<(unsigned)(((size_t)TOKS * DIMF + 255) / 256), 256, 0, stream>>>(H, AGG, gcnB[i]);
    // score + descending argsort
    gemv1<0><<<(TOKS + 7) / 8, 256, 0, stream>>>(H, scW[i], scB[i], SCORE, TOKS, 256);
    sort_init<<<(BATCH * SORTP + 255) / 256, 256, 0, stream>>>(SCORE, KEYS, IDX);
    for (int k = 2; k <= SORTP; k <<= 1)
      for (int j = k >> 1; j > 0; j >>= 1)
        bitonic_step<<<(BATCH * SORTP + 255) / 256, 256, 0, stream>>>(KEYS, IDX, j, k);
    // gather -> per-bin performer -> scatter back -> full performer
    gather_kernel<<<(unsigned)(((size_t)TOKS * 64 + 255) / 256), 256, 0, stream>>>(H, IDX, XS);
    run_performer(XS, BATCH * BINSN, G_GENES / BINSN, bperf[i],
                  HN, BQ, BK, BV, BATT, TMP1, KVb, KSb, stream);
    scatter_kernel<<<(unsigned)(((size_t)TOKS * 64 + 255) / 256), 256, 0, stream>>>(XS, IDX, H);
    run_performer(H, BATCH, G_GENES, fperf[i],
                  HN, BQ, BK, BV, BATT, TMP1, KVb, KSb, stream);
  }

  // ---- head: ln -> relu(h1) -> relu(h2)
  ln_kernel<<<(TOKS + 7) / 8, 256, 0, stream>>>(H, F(15), F(16), H, TOKS);
  gemm_launch(H, h1t, F(18), nullptr, TMP1, TOKS, 1024, 256, 1, stream);
  gemv1<1><<<(TOKS + 7) / 8, 256, 0, stream>>>(TMP1, F(19), F(20), (float*)d_out, TOKS, 1024);
}
